// TransformerModel_9698036154653
// MI455X (gfx1250) — matrix operand order repaired and checked by reading, not compiled, not run on hardware
//
#include <hip/hip_runtime.h>
#include <hip/hip_bf16.h>
#include <stdint.h>
#include <stddef.h>

// Model dims (compile-time constants from the reference)
#define B_    4
#define S_    512
#define D_    1024
#define H_    16
#define L_    6
#define DFF_  4096
#define V_    32000
#define DK_   64

typedef __bf16        v16bf __attribute__((ext_vector_type(16)));
typedef float         v8f   __attribute__((ext_vector_type(8)));
typedef unsigned int  u32x4 __attribute__((ext_vector_type(4)));

__device__ __forceinline__ unsigned short f2bf(float f) {
  unsigned int u = __float_as_uint(f);
  u += 0x7fffu + ((u >> 16) & 1u);   // round-to-nearest-even
  return (unsigned short)(u >> 16);
}

union ABFrag { v16bf v; u32x4 q[2]; };

// One k-step worth of operands: 2 A fragments (32 rows) x 4 B fragments (64 cols)
struct Frag6 { ABFrag a0, a1, b0, b1, b2, b3; };

__device__ __forceinline__ void load_frags(Frag6& f,
    const unsigned short* ar0, const unsigned short* ar1,
    const unsigned short* bp0, const unsigned short* bp1,
    const unsigned short* bp2, const unsigned short* bp3, int k0) {
  // A fragment: 16 contiguous K values per lane (half-wave K split)
  f.a0.q[0] = *(const u32x4*)(ar0 + k0);
  f.a0.q[1] = *(const u32x4*)(ar0 + k0 + 8);
  f.a1.q[0] = *(const u32x4*)(ar1 + k0);
  f.a1.q[1] = *(const u32x4*)(ar1 + k0 + 8);
  // B fragment: 16 contiguous K values per lane (half-wave K split)
  f.b0.q[0] = *(const u32x4*)(bp0 + k0);
  f.b0.q[1] = *(const u32x4*)(bp0 + k0 + 8);
  f.b1.q[0] = *(const u32x4*)(bp1 + k0);
  f.b1.q[1] = *(const u32x4*)(bp1 + k0 + 8);
  f.b2.q[0] = *(const u32x4*)(bp2 + k0);
  f.b2.q[1] = *(const u32x4*)(bp2 + k0 + 8);
  f.b3.q[0] = *(const u32x4*)(bp3 + k0);
  f.b3.q[1] = *(const u32x4*)(bp3 + k0 + 8);
}

__device__ __forceinline__ void wmma8(const Frag6& f, v8f acc[2][4]) {
  acc[0][0] = __builtin_amdgcn_wmma_f32_16x16x32_bf16(false, f.a0.v, false, f.b0.v,
                                                      (short)0, acc[0][0], false, false);
  acc[0][1] = __builtin_amdgcn_wmma_f32_16x16x32_bf16(false, f.a0.v, false, f.b1.v,
                                                      (short)0, acc[0][1], false, false);
  acc[0][2] = __builtin_amdgcn_wmma_f32_16x16x32_bf16(false, f.a0.v, false, f.b2.v,
                                                      (short)0, acc[0][2], false, false);
  acc[0][3] = __builtin_amdgcn_wmma_f32_16x16x32_bf16(false, f.a0.v, false, f.b3.v,
                                                      (short)0, acc[0][3], false, false);
  acc[1][0] = __builtin_amdgcn_wmma_f32_16x16x32_bf16(false, f.a1.v, false, f.b0.v,
                                                      (short)0, acc[1][0], false, false);
  acc[1][1] = __builtin_amdgcn_wmma_f32_16x16x32_bf16(false, f.a1.v, false, f.b1.v,
                                                      (short)0, acc[1][1], false, false);
  acc[1][2] = __builtin_amdgcn_wmma_f32_16x16x32_bf16(false, f.a1.v, false, f.b2.v,
                                                      (short)0, acc[1][2], false, false);
  acc[1][3] = __builtin_amdgcn_wmma_f32_16x16x32_bf16(false, f.a1.v, false, f.b3.v,
                                                      (short)0, acc[1][3], false, false);
}

enum { GF_F32 = 1, GF_BF16 = 2, GF_RELU = 4, GF_BIAS = 8 };

struct GemmP {
  const unsigned short* A;  long long sAb, sAh; int lda;
  const unsigned short* Bt; long long sBb, sBh; int ldb;
  float* Cf; unsigned short* Cb; long long sCb, sCh; int ldc;
  const float* bias;
  int M, N, K, Hh, flags;
  float scale;
};

// ---------------------------------------------------------------------------
// Batched GEMM: C[M,N] = scale * (A[M,K]_bf16 @ Bt[N,K]_bf16^T) (+bias) (ReLU)
// Wave tile 32x64 (8 accumulators, 8 WMMAs per k-step), 8 waves/block stacked
// in M -> 256x64 block tile. k-loop unrolled 2x with ping-pong fragment sets:
// no register copies, no in-loop branches, loads of one stage overlap the
// WMMAs of the other. Requires K % 64 == 0 (true for all GEMMs here).
// ---------------------------------------------------------------------------
__global__ __launch_bounds__(256) void gemm_bf16_wmma(GemmP p) {
  const int lane = threadIdx.x & 31;
  const int wv   = threadIdx.x >> 5;
  const int z  = blockIdx.z;
  const int bb = z / p.Hh;
  const int hh = z % p.Hh;
  const unsigned short* A  = p.A  + (long long)bb * p.sAb + (long long)hh * p.sAh;
  const unsigned short* Bt = p.Bt + (long long)bb * p.sBb + (long long)hh * p.sBh;

  const int m0 = blockIdx.y * 256 + wv * 32;
  if (m0 >= p.M) return;                       // wave-uniform: EXEC stays all-ones
  const int n0   = blockIdx.x * 64;
  const int half = lane >> 4;
  const int l15  = lane & 15;

  const unsigned short* ar0 = A + (long long)(m0 +      l15) * p.lda + half * 16;
  const unsigned short* ar1 = A + (long long)(m0 + 16 + l15) * p.lda + half * 16;
  const unsigned short* bp0 = Bt + (long long)(n0 +  0 + l15) * p.ldb + half * 16;
  const unsigned short* bp1 = Bt + (long long)(n0 + 16 + l15) * p.ldb + half * 16;
  const unsigned short* bp2 = Bt + (long long)(n0 + 32 + l15) * p.ldb + half * 16;
  const unsigned short* bp3 = Bt + (long long)(n0 + 48 + l15) * p.ldb + half * 16;

  v8f acc[2][4];
#pragma unroll
  for (int mi = 0; mi < 2; ++mi)
#pragma unroll
    for (int ni = 0; ni < 4; ++ni) acc[mi][ni] = (v8f){0,0,0,0,0,0,0,0};

  Frag6 f0, f1;
  load_frags(f0, ar0, ar1, bp0, bp1, bp2, bp3, 0);

  int k0 = 0;
  for (; k0 + 64 < p.K; k0 += 64) {
    load_frags(f1, ar0, ar1, bp0, bp1, bp2, bp3, k0 + 32);
    __builtin_prefetch(ar0 + k0 + 128, 0, 3);
    __builtin_prefetch(ar1 + k0 + 128, 0, 3);
    __builtin_prefetch(bp0 + k0 + 128, 0, 3);
    wmma8(f0, acc);
    load_frags(f0, ar0, ar1, bp0, bp1, bp2, bp3, k0 + 64);
    wmma8(f1, acc);
  }
  // tail: exactly 64 K remaining (K % 64 == 0)
  load_frags(f1, ar0, ar1, bp0, bp1, bp2, bp3, k0 + 32);
  wmma8(f0, acc);
  wmma8(f1, acc);

  float bi[4] = {0.f, 0.f, 0.f, 0.f};
  if (p.flags & GF_BIAS) {
#pragma unroll
    for (int ni = 0; ni < 4; ++ni) bi[ni] = p.bias[n0 + ni * 16 + l15];
  }
  const long long cz = (long long)bb * p.sCb + (long long)hh * p.sCh;
#pragma unroll
  for (int mi = 0; mi < 2; ++mi) {
#pragma unroll
    for (int r = 0; r < 8; ++r) {
      const int row = m0 + mi * 16 + half * 8 + r;
      const long long cb = cz + (long long)row * p.ldc + n0 + l15;
#pragma unroll
      for (int ni = 0; ni < 4; ++ni) {
        float v = acc[mi][ni][r] * p.scale + bi[ni];
        if (p.flags & GF_RELU) v = fmaxf(v, 0.f);
        if (p.flags & GF_F32)  p.Cf[cb + ni * 16] = v;
        if (p.flags & GF_BF16) p.Cb[cb + ni * 16] = f2bf(v);
      }
    }
  }
}

// ---------------------------------------------------------------------------
// Masked softmax over scores[B,H,S,S] (already scaled) -> P bf16.
// causal=0: key valid iff tok[b,k]!=0 (src padding mask)
// causal=1: valid iff tok[b,q]!=0 && k<=q (decoder self-attn)
// Matches reference: invalid positions become -1e9 before softmax.
// ---------------------------------------------------------------------------
__global__ __launch_bounds__(128) void softmax_kernel(const float* Sc,
                                                      unsigned short* P,
                                                      const int* tok, int causal) {
  const int q = blockIdx.x, h = blockIdx.y, b = blockIdx.z;
  const int t = threadIdx.x;
  const long long rb = ((((long long)b * H_ + h) * S_) + q) * S_;
  __shared__ float red[128];

  float v[4];
#pragma unroll
  for (int j = 0; j < 4; ++j) {
    const int k = t + j * 128;
    bool valid = causal ? ((tok[b * S_ + q] != 0) && (k <= q))
                        : (tok[b * S_ + k] != 0);
    v[j] = valid ? Sc[rb + k] : -1e9f;
  }
  float mx = fmaxf(fmaxf(v[0], v[1]), fmaxf(v[2], v[3]));
  red[t] = mx; __syncthreads();
  for (int s = 64; s > 0; s >>= 1) {
    if (t < s) red[t] = fmaxf(red[t], red[t + s]);
    __syncthreads();
  }
  mx = red[0]; __syncthreads();

  float e[4], sum = 0.f;
#pragma unroll
  for (int j = 0; j < 4; ++j) { e[j] = __expf(v[j] - mx); sum += e[j]; }
  red[t] = sum; __syncthreads();
  for (int s = 64; s > 0; s >>= 1) {
    if (t < s) red[t] += red[t + s];
    __syncthreads();
  }
  const float inv = 1.f / red[0];
#pragma unroll
  for (int j = 0; j < 4; ++j) P[rb + t + j * 128] = f2bf(e[j] * inv);
}

// ---------------------------------------------------------------------------
// h = LayerNorm(x + a) * g + b ; writes fp32 (residual stream) + bf16 (GEMM A)
// ---------------------------------------------------------------------------
__global__ __launch_bounds__(256) void add_ln_kernel(const float* x, const float* a,
                                                     const float* g, const float* be,
                                                     float* of, unsigned short* ob) {
  const long long rb = (long long)blockIdx.x * D_;
  const int t = threadIdx.x;
  __shared__ float r1[256], r2[256];

  float v[4]; float s = 0.f, s2 = 0.f;
#pragma unroll
  for (int j = 0; j < 4; ++j) {
    const int i = t + j * 256;
    v[j] = x[rb + i] + a[rb + i];
    s += v[j]; s2 += v[j] * v[j];
  }
  r1[t] = s; r2[t] = s2; __syncthreads();
  for (int st = 128; st > 0; st >>= 1) {
    if (t < st) { r1[t] += r1[t + st]; r2[t] += r2[t + st]; }
    __syncthreads();
  }
  const float m   = r1[0] * (1.f / D_);
  const float var = r2[0] * (1.f / D_) - m * m;
  const float inv = rsqrtf(var + 1e-5f);
#pragma unroll
  for (int j = 0; j < 4; ++j) {
    const int i = t + j * 256;
    const float y = (v[j] - m) * inv * g[i] + be[i];
    of[rb + i] = y;
    ob[rb + i] = f2bf(y);
  }
}

// x[b,s,:] = emb[tok[b,s]] + pe[s,:]   (fp32 + bf16 outputs)
__global__ __launch_bounds__(256) void embed_kernel(const int* tok, const float* emb,
                                                    const float* pe, float* of,
                                                    unsigned short* ob) {
  const int row = blockIdx.x;              // b*S + s
  const int s = row % S_;
  const long long eb = (long long)tok[row] * D_;
  const long long rb = (long long)row * D_;
  for (int i = threadIdx.x; i < D_; i += 256) {
    const float v = emb[eb + i] + pe[(long long)s * D_ + i];
    of[rb + i] = v;
    ob[rb + i] = f2bf(v);
  }
}

// fp32 [R,C] -> bf16 [C,R] (LDS-tiled transpose+convert; R,C multiples of 32)
__global__ __launch_bounds__(256) void tconv_kernel(const float* in,
                                                    unsigned short* out,
                                                    int R, int C) {
  __shared__ float tile[32][33];
  const int tx = threadIdx.x, ty = threadIdx.y;   // (32, 8)
  const int c0 = blockIdx.x * 32, r0 = blockIdx.y * 32;
#pragma unroll
  for (int j = 0; j < 4; ++j)
    tile[ty + j * 8][tx] = in[(long long)(r0 + ty + j * 8) * C + c0 + tx];
  __syncthreads();
#pragma unroll
  for (int j = 0; j < 4; ++j)
    out[(long long)(c0 + ty + j * 8) * R + r0 + tx] = f2bf(tile[tx][ty + j * 8]);
}

// Vt[b,h,d,s] = V[b,s,h*DK+d]  (bf16 head-major transpose, LDS tiled so both
// global read and write are coalesced)
__global__ __launch_bounds__(256) void vheadT_kernel(const unsigned short* Vb,
                                                     unsigned short* Vt) {
  __shared__ unsigned short tile[32][33];
  const int tx = threadIdx.x, ty = threadIdx.y;   // (32, 8)
  const int z = blockIdx.z;
  const int b = z / H_, h = z % H_;
  const int s0 = blockIdx.x * 32, d0 = blockIdx.y * 32;
#pragma unroll
  for (int j = 0; j < 4; ++j)
    tile[ty + j * 8][tx] =
        Vb[((long long)(b * S_ + s0 + ty + j * 8)) * D_ + h * DK_ + d0 + tx];
  __syncthreads();
#pragma unroll
  for (int j = 0; j < 4; ++j)
    Vt[((long long)((b * H_ + h) * DK_ + d0 + ty + j * 8)) * S_ + s0 + tx] =
        tile[tx][ty + j * 8];
}

// ---------------------------------------------------------------------------
extern "C" void kernel_launch(void* const* d_in, const int* in_sizes, int n_in,
                              void* d_out, int out_size, void* d_ws, size_t ws_size,
                              hipStream_t stream) {
  (void)in_sizes; (void)n_in; (void)out_size; (void)ws_size;

  const int*   src     = (const int*)  d_in[0];
  const int*   tgt     = (const int*)  d_in[1];
  const float* enc_emb = (const float*)d_in[2];
  const float* dec_emb = (const float*)d_in[3];
  const float* pe      = (const float*)d_in[4];
  const float* EP[16]; for (int i = 0; i < 16; ++i) EP[i] = (const float*)d_in[5 + i];
  const float* DP[26]; for (int i = 0; i < 26; ++i) DP[i] = (const float*)d_in[21 + i];
  const float* Wout = (const float*)d_in[47];
  const float* bout = (const float*)d_in[48];

  // ---- bump allocator over workspace ----
  const uintptr_t base = (uintptr_t)d_ws;
  size_t off = 0;
  auto alloc = [&](size_t bytes) -> void* {
    off = (off + 255) & ~(size_t)255;
    void* p = (void*)(base + off);
    off += bytes;
    return p;
  };

  auto tconv = [&](const float* in, unsigned short* out, int R, int C) {
    tconv_kernel<<<dim3(C / 32, R / 32), dim3(32, 8), 0, stream>>>(in, out, R, C);
  };

  // ---- convert all weights to bf16, transposed [N,K] ----
  unsigned short* WoutT = (unsigned short*)alloc((size_t)D_ * V_ * 2);
  tconv(Wout, WoutT, D_, V_);

  unsigned short *eWqT[L_], *eWkT[L_], *eWvT[L_], *eWoT[L_], *eW1T[L_], *eW2T[L_];
  unsigned short *dWqT[L_], *dWkT[L_], *dWvT[L_], *dWoT[L_], *dW1T[L_], *dW2T[L_];
  unsigned short *cWqT[L_], *cWkT[L_], *cWvT[L_], *cWoT[L_];
  for (int l = 0; l < L_; ++l) {
    const size_t oDD  = (size_t)l * D_ * D_;
    const size_t oDF  = (size_t)l * D_ * DFF_;
    eWqT[l] = (unsigned short*)alloc((size_t)D_ * D_ * 2);  tconv(EP[0]  + oDD, eWqT[l], D_, D_);
    eWkT[l] = (unsigned short*)alloc((size_t)D_ * D_ * 2);  tconv(EP[2]  + oDD, eWkT[l], D_, D_);
    eWvT[l] = (unsigned short*)alloc((size_t)D_ * D_ * 2);  tconv(EP[4]  + oDD, eWvT[l], D_, D_);
    eWoT[l] = (unsigned short*)alloc((size_t)D_ * D_ * 2);  tconv(EP[6]  + oDD, eWoT[l], D_, D_);
    eW1T[l] = (unsigned short*)alloc((size_t)D_ * DFF_ * 2); tconv(EP[8]  + oDF, eW1T[l], D_, DFF_);
    eW2T[l] = (unsigned short*)alloc((size_t)D_ * DFF_ * 2); tconv(EP[10] + oDF, eW2T[l], DFF_, D_);

    dWqT[l] = (unsigned short*)alloc((size_t)D_ * D_ * 2);  tconv(DP[0]  + oDD, dWqT[l], D_, D_);
    dWkT[l] = (unsigned short*)alloc((size_t)D_ * D_ * 2);  tconv(DP[2]  + oDD, dWkT[l], D_, D_);
    dWvT[l] = (unsigned short*)alloc((size_t)D_ * D_ * 2);  tconv(DP[4]  + oDD, dWvT[l], D_, D_);
    dWoT[l] = (unsigned short*)alloc((size_t)D_ * D_ * 2);  tconv(DP[6]  + oDD, dWoT[l], D_, D_);
    dW1T[l] = (unsigned short*)alloc((size_t)D_ * DFF_ * 2); tconv(DP[8]  + oDF, dW1T[l], D_, DFF_);
    dW2T[l] = (unsigned short*)alloc((size_t)D_ * DFF_ * 2); tconv(DP[10] + oDF, dW2T[l], DFF_, D_);
    cWqT[l] = (unsigned short*)alloc((size_t)D_ * D_ * 2);  tconv(DP[16] + oDD, cWqT[l], D_, D_);
    cWkT[l] = (unsigned short*)alloc((size_t)D_ * D_ * 2);  tconv(DP[18] + oDD, cWkT[l], D_, D_);
    cWvT[l] = (unsigned short*)alloc((size_t)D_ * D_ * 2);  tconv(DP[20] + oDD, cWvT[l], D_, D_);
    cWoT[l] = (unsigned short*)alloc((size_t)D_ * D_ * 2);  tconv(DP[22] + oDD, cWoT[l], D_, D_);
  }

  // ---- activations ----
  const size_t nBSD = (size_t)B_ * S_ * D_;
  float*          xf     = (float*)alloc(nBSD * 4);
  unsigned short* xb     = (unsigned short*)alloc(nBSD * 2);
  float*          yf     = (float*)alloc(nBSD * 4);
  unsigned short* yb     = (unsigned short*)alloc(nBSD * 2);
  unsigned short* qb     = (unsigned short*)alloc(nBSD * 2);
  unsigned short* kb     = (unsigned short*)alloc(nBSD * 2);
  unsigned short* vb     = (unsigned short*)alloc(nBSD * 2);
  unsigned short* vtb    = (unsigned short*)alloc(nBSD * 2);
  unsigned short* attnb  = (unsigned short*)alloc(nBSD * 2);
  float*          af     = (float*)alloc(nBSD * 4);
  float*          scores = (float*)alloc((size_t)B_ * H_ * S_ * S_ * 4);
  unsigned short* pb     = (unsigned short*)alloc((size_t)B_ * H_ * S_ * S_ * 2);
  unsigned short* f1b    = (unsigned short*)alloc((size_t)B_ * S_ * DFF_ * 2);

  // ---- GEMM launcher (block tile 256x64) ----
  auto gemm = [&](const unsigned short* A, long long sAb, long long sAh, int lda,
                  const unsigned short* Bt, long long sBb, long long sBh, int ldb,
                  float* Cf, unsigned short* Cb, long long sCb, long long sCh, int ldc,
                  const float* bias, int M, int N, int K, float scale, int flags,
                  int Zb, int Zh) {
    GemmP p;
    p.A = A;  p.sAb = sAb; p.sAh = sAh; p.lda = lda;
    p.Bt = Bt; p.sBb = sBb; p.sBh = sBh; p.ldb = ldb;
    p.Cf = Cf; p.Cb = Cb; p.sCb = sCb; p.sCh = sCh; p.ldc = ldc;
    p.bias = bias; p.M = M; p.N = N; p.K = K; p.Hh = Zh;
    p.flags = flags; p.scale = scale;
    dim3 grid(N / 64, (M + 255) / 256, Zb * Zh);
    gemm_bf16_wmma<<<grid, 256, 0, stream>>>(p);
  };

  auto addln = [&](float* x, const float* a, const float* g, const float* be,
                   float* of, unsigned short* ob) {
    add_ln_kernel<<<B_ * S_, 256, 0, stream>>>(x, a, g, be, of, ob);
  };

  const int M = B_ * S_;

  // ---- full MHA block: out (fp32 [B,S,D]) = MHA(inQ, inKV) ----
  auto attention = [&](const unsigned short* inQ, const unsigned short* inKV,
                       const unsigned short* WqT, const float* bq,
                       const unsigned short* WkT, const float* bk,
                       const unsigned short* WvT, const float* bv,
                       const unsigned short* WoT, const float* bo,
                       const int* mtok, int causal, float* outA) {
    gemm(inQ, 0, 0, D_, WqT, 0, 0, D_, nullptr, qb, 0, 0, D_, bq,
         M, D_, D_, 1.f, GF_BF16 | GF_BIAS, 1, 1);
    gemm(inKV, 0, 0, D_, WkT, 0, 0, D_, nullptr, kb, 0, 0, D_, bk,
         M, D_, D_, 1.f, GF_BF16 | GF_BIAS, 1, 1);
    gemm(inKV, 0, 0, D_, WvT, 0, 0, D_, nullptr, vb, 0, 0, D_, bv,
         M, D_, D_, 1.f, GF_BF16 | GF_BIAS, 1, 1);
    vheadT_kernel<<<dim3(S_ / 32, DK_ / 32, B_ * H_), dim3(32, 8), 0, stream>>>(vb, vtb);
    // scores[b,h] = Q_h K_h^T / sqrt(DK)
    gemm(qb, (long long)S_ * D_, DK_, D_,
         kb, (long long)S_ * D_, DK_, D_,
         scores, nullptr, (long long)H_ * S_ * S_, (long long)S_ * S_, S_,
         nullptr, S_, S_, DK_, 0.125f, GF_F32, B_, H_);
    softmax_kernel<<<dim3(S_, H_, B_), 128, 0, stream>>>(scores, pb, mtok, causal);
    // attn[b,s,h*DK+d] = P V
    gemm(pb, (long long)H_ * S_ * S_, (long long)S_ * S_, S_,
         vtb, (long long)H_ * DK_ * S_, (long long)DK_ * S_, S_,
         nullptr, attnb, (long long)S_ * D_, DK_, D_,
         nullptr, S_, DK_, S_, 1.f, GF_BF16, B_, H_);
    gemm(attnb, 0, 0, D_, WoT, 0, 0, D_, outA, nullptr, 0, 0, D_, bo,
         M, D_, D_, 1.f, GF_F32 | GF_BIAS, 1, 1);
  };

  // ================= Encoder =================
  embed_kernel<<<B_ * S_, 256, 0, stream>>>(src, enc_emb, pe, xf, xb);
  for (int l = 0; l < L_; ++l) {
    const size_t oD = (size_t)l * D_, oF = (size_t)l * DFF_;
    attention(xb, xb, eWqT[l], EP[1] + oD, eWkT[l], EP[3] + oD,
              eWvT[l], EP[5] + oD, eWoT[l], EP[7] + oD, src, 0, af);
    addln(xf, af, EP[12] + oD, EP[13] + oD, xf, xb);
    gemm(xb, 0, 0, D_, eW1T[l], 0, 0, D_, nullptr, f1b, 0, 0, DFF_, EP[9] + oF,
         M, DFF_, D_, 1.f, GF_BF16 | GF_BIAS | GF_RELU, 1, 1);
    gemm(f1b, 0, 0, DFF_, eW2T[l], 0, 0, DFF_, af, nullptr, 0, 0, D_, EP[11] + oD,
         M, D_, DFF_, 1.f, GF_F32 | GF_BIAS, 1, 1);
    addln(xf, af, EP[14] + oD, EP[15] + oD, xf, xb);
  }
  // xb now holds encoder output (bf16) for cross-attention K/V.

  // ================= Decoder =================
  embed_kernel<<<B_ * S_, 256, 0, stream>>>(tgt, dec_emb, pe, yf, yb);
  for (int l = 0; l < L_; ++l) {
    const size_t oD = (size_t)l * D_, oF = (size_t)l * DFF_;
    // masked self-attention
    attention(yb, yb, dWqT[l], DP[1] + oD, dWkT[l], DP[3] + oD,
              dWvT[l], DP[5] + oD, dWoT[l], DP[7] + oD, tgt, 1, af);
    addln(yf, af, DP[12] + oD, DP[13] + oD, yf, yb);
    // cross-attention (K/V from encoder output, src-pad mask)
    attention(yb, xb, cWqT[l], DP[17] + oD, cWkT[l], DP[19] + oD,
              cWvT[l], DP[21] + oD, cWoT[l], DP[23] + oD, src, 0, af);
    addln(yf, af, DP[14] + oD, DP[15] + oD, yf, yb);
    // FFN
    gemm(yb, 0, 0, D_, dW1T[l], 0, 0, D_, nullptr, f1b, 0, 0, DFF_, DP[9] + oF,
         M, DFF_, D_, 1.f, GF_BF16 | GF_BIAS | GF_RELU, 1, 1);
    gemm(f1b, 0, 0, DFF_, dW2T[l], 0, 0, DFF_, af, nullptr, 0, 0, D_, DP[11] + oD,
         M, D_, DFF_, 1.f, GF_F32 | GF_BIAS, 1, 1);
    addln(yf, af, DP[24] + oD, DP[25] + oD, yf, yb);
  }

  // ================= Output projection =================
  gemm(yb, 0, 0, D_, WoutT, 0, 0, D_, (float*)d_out, nullptr, 0, 0, V_, bout,
       M, V_, D_, 1.f, GF_F32 | GF_BIAS, 1, 1);
}
